// EncodingLayer_45019847197063
// MI455X (gfx1250) — matrix operand order repaired and checked by reading, not compiled, not run on hardware
//
#include <hip/hip_runtime.h>

typedef __attribute__((ext_vector_type(8)))  __bf16 v8bf;
typedef __attribute__((ext_vector_type(16))) __bf16 v16bf;
typedef __attribute__((ext_vector_type(8)))  float  v8f;

static constexpr int Bv = 2;
static constexpr int Cv = 64;
static constexpr int Nv = 40000;
static constexpr int Kn = 6;
static constexpr int S  = Bv * Nv * Kn;        // 480000 sample rows
static constexpr int MT = 5;                   // 16-row m-tiles per wave
static constexpr int WAVES = 8;                // waves per workgroup (256 thr, wave32)
static constexpr int NWG = S / (WAVES * MT * 16);  // 750 exactly
static constexpr float EPSf = 1e-5f;

// ---------------- small helpers ----------------
__global__ void cvt_bf16(const float* __restrict__ src, __bf16* __restrict__ dst, int n) {
  int i = blockIdx.x * blockDim.x + threadIdx.x;
  if (i < n) dst[i] = (__bf16)src[i];
}

// points (B,C,N) f32 -> xT (B*N, 64) bf16, LDS-tiled transpose
__launch_bounds__(256)
__global__ void transpose_pts(const float* __restrict__ pts, __bf16* __restrict__ xT) {
  __shared__ float tile[64][33];
  const int blk = blockIdx.x;
  const int b   = blk / (Nv / 32);
  const int n0  = (blk % (Nv / 32)) * 32;
  const int nn  = threadIdx.x & 31;
  const int cg  = threadIdx.x >> 5;            // 8 groups of 8 channels
  const float* p = pts + (size_t)b * Cv * Nv;
  #pragma unroll
  for (int i = 0; i < 8; ++i) {
    const int c = cg * 8 + i;
    tile[c][nn] = p[(size_t)c * Nv + n0 + nn]; // coalesced 128B per row
  }
  __syncthreads();
  const int r  = threadIdx.x >> 3;             // 0..31 point within tile
  const int cp = threadIdx.x & 7;              // 8 chunks of 8 channels
  v8bf o;
  #pragma unroll
  for (int i = 0; i < 8; ++i) o[i] = (__bf16)tile[cp * 8 + i][r];
  *(v8bf*)(xT + (size_t)(b * Nv + n0 + r) * 64 + cp * 8) = o;  // coalesced 16B
}

// ---------------- fused GEMM + bias + BN-stat + (optional gather / input-norm) ----------------
template<int CIN, int COUT, bool GATHER, bool NORM>
__launch_bounds__(256)
__global__ void gemm_bn(const __bf16* A,                      // !GATHER: S x CIN (pre-norm)
                        const __bf16* __restrict__ xT,        // GATHER source
                        const int*    __restrict__ nearIdx,
                        const __bf16* __restrict__ Wb,        // COUT x CIN bf16
                        const float*  __restrict__ bias,
                        const float*  __restrict__ scale,     // CIN (NORM)
                        const float*  __restrict__ shift,     // CIN (NORM)
                        __bf16* Y,                            // S x COUT bf16 (pre-norm)
                        float* __restrict__ partials)         // NWG x 2 x COUT
{
  constexpr int NT = COUT / 16;
  constexpr int KC = CIN / 32;
  __shared__ __bf16 stg[WAVES * 256];
  __shared__ float wsum[WAVES][COUT];
  __shared__ float wsq [WAVES][COUT];
  __shared__ float lsc[CIN], lsh[CIN];

  const int tid  = threadIdx.x;
  const int wave = tid >> 5;
  const int lane = tid & 31;
  const int lrow = lane & 15;
  const int lhi  = lane >> 4;                  // which K-half this lane owns

  if (NORM) {
    if (tid < CIN) { lsc[tid] = scale[tid]; lsh[tid] = shift[tid]; }
    __syncthreads();
  }

  float biasv[NT];
  #pragma unroll
  for (int nt = 0; nt < NT; ++nt) biasv[nt] = bias[nt * 16 + lrow];

  float sumAcc[NT], sqAcc[NT];
  #pragma unroll
  for (int nt = 0; nt < NT; ++nt) { sumAcc[nt] = 0.f; sqAcc[nt] = 0.f; }

  v8bf pad8;
  #pragma unroll
  for (int i = 0; i < 8; ++i) pad8[i] = (__bf16)(-1.0f);

  const int mbase0 = (blockIdx.x * WAVES + wave) * (MT * 16);

  for (int mt = 0; mt < MT; ++mt) {
    const int mbase = mbase0 + mt * 16;
    const int row   = mbase + lrow;

    // ---- A fragments: 16x32 bf16 per K-chunk; lane holds K {h*8..h*8+7, h*8+16..h*8+23}
    v16bf a[KC];
    if (GATHER) {
      const int kidx = row % Kn;
      const int t    = row / Kn;
      const int pn   = t % Nv;
      const int pb   = t / Nv;
      const __bf16* ctr = xT + (size_t)(pb * Nv + pn) * 64;
      const int idx     = nearIdx[pn * Kn + kidx];
      const bool valid  = (idx < Nv);
      const __bf16* nbr = xT + (size_t)(pb * Nv + (valid ? idx : 0)) * 64;
      #pragma unroll
      for (int kc = 0; kc < KC; ++kc) {
        const int o = (kc & 1) * 32 + lhi * 8;
        v8bf lo, hi;
        if (kc < 2) {                       // channels 0..63 : center point
          lo = *(const v8bf*)(ctr + o);
          hi = *(const v8bf*)(ctr + o + 16);
        } else if (valid) {                 // channels 64..127 : neighbor (or -1 pad)
          lo = *(const v8bf*)(nbr + o);
          hi = *(const v8bf*)(nbr + o + 16);
        } else { lo = pad8; hi = pad8; }
        #pragma unroll
        for (int i = 0; i < 8; ++i) { a[kc][i] = lo[i]; a[kc][8 + i] = hi[i]; }
      }
    } else {
      const __bf16* ap = A + (size_t)row * CIN;
      #pragma unroll
      for (int kc = 0; kc < KC; ++kc) {
        const int c0 = kc * 32 + lhi * 8;
        v8bf lo = *(const v8bf*)(ap + c0);
        v8bf hi = *(const v8bf*)(ap + c0 + 16);
        if (NORM) {                          // fused BN-apply + ReLU on operand load
          #pragma unroll
          for (int i = 0; i < 8; ++i) {
            float f0 = fmaxf(fmaf((float)lo[i], lsc[c0 + i],      lsh[c0 + i]),      0.f);
            float f1 = fmaxf(fmaf((float)hi[i], lsc[c0 + 16 + i], lsh[c0 + 16 + i]), 0.f);
            a[kc][i]     = (__bf16)f0;
            a[kc][8 + i] = (__bf16)f1;
          }
        } else {
          #pragma unroll
          for (int i = 0; i < 8; ++i) { a[kc][i] = lo[i]; a[kc][8 + i] = hi[i]; }
        }
      }
    }

    // ---- per output-channel tile: WMMA chain over K, bias in accumulator
    #pragma unroll
    for (int nt = 0; nt < NT; ++nt) {
      v8f c;
      #pragma unroll
      for (int i = 0; i < 8; ++i) c[i] = biasv[nt];
      const __bf16* wp = Wb + (size_t)(nt * 16 + lrow) * CIN + lhi * 8;  // B col = W row
      #pragma unroll
      for (int kc = 0; kc < KC; ++kc) {
        v8bf blo = *(const v8bf*)(wp + kc * 32);
        v8bf bhi = *(const v8bf*)(wp + kc * 32 + 16);
        v16bf bfrag;
        #pragma unroll
        for (int i = 0; i < 8; ++i) { bfrag[i] = blo[i]; bfrag[8 + i] = bhi[i]; }
        c = __builtin_amdgcn_wmma_f32_16x16x32_bf16(false, a[kc], false, bfrag,
                                                    (short)0, c, false, false);
      }
      // BN statistics from fp32 accumulators (lane owns one column)
      float s = 0.f, q = 0.f;
      #pragma unroll
      for (int i = 0; i < 8; ++i) { s += c[i]; q += c[i] * c[i]; }
      sumAcc[nt] += s; sqAcc[nt] += q;

      // stage 16x16 tile in LDS -> coalesced 16B global stores
      __bf16* st = stg + wave * 256;
      #pragma unroll
      for (int i = 0; i < 8; ++i) st[(lhi * 8 + i) * 16 + lrow] = (__bf16)c[i];
      asm volatile("s_wait_dscnt 0" ::: "memory");
      v8bf ov = *(const v8bf*)(st + (lane >> 1) * 16 + (lane & 1) * 8);
      *(v8bf*)(Y + (size_t)(mbase + (lane >> 1)) * COUT + nt * 16 + (lane & 1) * 8) = ov;
    }
  }

  // combine the two lane-halves of each column, then fixed-order WG reduction (deterministic)
  #pragma unroll
  for (int nt = 0; nt < NT; ++nt) {
    float s = sumAcc[nt] + __shfl_xor(sumAcc[nt], 16, 32);
    float q = sqAcc[nt]  + __shfl_xor(sqAcc[nt],  16, 32);
    if (lane < 16) { wsum[wave][nt * 16 + lrow] = s; wsq[wave][nt * 16 + lrow] = q; }
  }
  __syncthreads();
  if (tid < COUT) {
    float s = 0.f, q = 0.f;
    #pragma unroll
    for (int w = 0; w < WAVES; ++w) { s += wsum[w][tid]; q += wsq[w][tid]; }
    partials[(size_t)blockIdx.x * 2 * COUT + tid]        = s;
    partials[(size_t)blockIdx.x * 2 * COUT + COUT + tid] = q;
  }
}

// fold mean/var + gamma/beta into per-channel scale/shift
__global__ void reduce_stats(const float* __restrict__ partials, int nwg, int cout, float invM,
                             const float* __restrict__ g, const float* __restrict__ beta,
                             float* __restrict__ sc, float* __restrict__ sh) {
  const int c = threadIdx.x;
  if (c >= cout) return;
  float s = 0.f, q = 0.f;
  for (int w = 0; w < nwg; ++w) {
    s += partials[(size_t)w * 2 * cout + c];
    q += partials[(size_t)w * 2 * cout + cout + c];
  }
  const float mu  = s * invM;
  const float var = q * invM - mu * mu;
  const float r   = rsqrtf(var + EPSf);
  const float sfc = g[c] * r;
  sc[c] = sfc;
  sh[c] = beta[c] - mu * sfc;
}

// BN2 + ReLU + max over 6 neighbors -> new_feature (B,128,N); max over channels -> out_feature (B,N)
__launch_bounds__(256)
__global__ void out_kernel(const __bf16* __restrict__ y2,
                           const float* __restrict__ sc, const float* __restrict__ sh,
                           float* __restrict__ out) {
  __shared__ float lsc[128], lsh[128];
  __shared__ float red[8][32];
  const int tid = threadIdx.x;
  if (tid < 128) { lsc[tid] = sc[tid]; lsh[tid] = sh[tid]; }
  __syncthreads();
  const int nblk = blockIdx.x % (Nv / 32);
  const int b    = blockIdx.x / (Nv / 32);
  const int nn   = tid & 31;
  const int cg   = tid >> 5;                   // 8 groups of 16 channels
  const int n    = nblk * 32 + nn;
  const int c0   = cg * 16;
  float m[16];
  #pragma unroll
  for (int j = 0; j < 16; ++j) m[j] = 0.f;     // ReLU outputs are >= 0
  for (int k = 0; k < Kn; ++k) {
    const __bf16* r = y2 + ((size_t)((b * Nv + n) * Kn + k)) * 128 + c0;
    v8bf lo = *(const v8bf*)r;
    v8bf hi = *(const v8bf*)(r + 8);
    #pragma unroll
    for (int j = 0; j < 8; ++j) {
      float v0 = fmaxf(fmaf((float)lo[j], lsc[c0 + j],     lsh[c0 + j]),     0.f);
      float v1 = fmaxf(fmaf((float)hi[j], lsc[c0 + 8 + j], lsh[c0 + 8 + j]), 0.f);
      m[j]     = fmaxf(m[j], v0);
      m[8 + j] = fmaxf(m[8 + j], v1);
    }
  }
  float* nf = out + (size_t)Bv * Nv + (size_t)b * 128 * Nv + n;  // new_feature region
  #pragma unroll
  for (int j = 0; j < 16; ++j) nf[(size_t)(c0 + j) * Nv] = m[j]; // 128B coalesced per j
  float pm = m[0];
  #pragma unroll
  for (int j = 1; j < 16; ++j) pm = fmaxf(pm, m[j]);
  red[cg][nn] = pm;
  __syncthreads();
  if (tid < 32) {
    float v = red[0][tid];
    #pragma unroll
    for (int w = 1; w < 8; ++w) v = fmaxf(v, red[w][tid]);
    out[(size_t)b * Nv + nblk * 32 + tid] = v;                   // out_feature region
  }
}

// ---------------- host side ----------------
extern "C" void kernel_launch(void* const* d_in, const int* in_sizes, int n_in,
                              void* d_out, int out_size, void* d_ws, size_t ws_size,
                              hipStream_t stream) {
  (void)in_sizes; (void)n_in; (void)out_size; (void)ws_size;
  const float* points  = (const float*)d_in[0];
  const int*   nearIdx = (const int*)d_in[1];
  const float* W0 = (const float*)d_in[2];
  const float* b0 = (const float*)d_in[3];
  const float* g0 = (const float*)d_in[4];
  const float* e0 = (const float*)d_in[5];
  const float* W1 = (const float*)d_in[6];
  const float* b1 = (const float*)d_in[7];
  const float* g1 = (const float*)d_in[8];
  const float* e1 = (const float*)d_in[9];
  const float* W2 = (const float*)d_in[10];
  const float* b2 = (const float*)d_in[11];
  const float* g2 = (const float*)d_in[12];
  const float* e2 = (const float*)d_in[13];
  float* out = (float*)d_out;

  char* ws = (char*)d_ws;
  size_t off = 0;
  auto alloc = [&](size_t bytes) -> char* {
    off = (off + 255) & ~(size_t)255;
    char* p = ws + off;
    off += bytes;
    return p;
  };
  __bf16* xT  = (__bf16*)alloc((size_t)Bv * Nv * 64 * 2);   // 10.24 MB
  __bf16* wb0 = (__bf16*)alloc((size_t)64 * 128 * 2);
  __bf16* wb1 = (__bf16*)alloc((size_t)64 * 64 * 2);
  __bf16* wb2 = (__bf16*)alloc((size_t)128 * 64 * 2);
  __bf16* y01 = (__bf16*)alloc((size_t)S * 64 * 2);         // y0, then y1 in place
  __bf16* y2  = (__bf16*)alloc((size_t)S * 128 * 2);
  float*  prt = (float*)alloc((size_t)NWG * 2 * 128 * 4);
  float*  sc0 = (float*)alloc(64 * 4);
  float*  sh0 = (float*)alloc(64 * 4);
  float*  sc1 = (float*)alloc(64 * 4);
  float*  sh1 = (float*)alloc(64 * 4);
  float*  sc2 = (float*)alloc(128 * 4);
  float*  sh2 = (float*)alloc(128 * 4);

  const float invM = 1.0f / (float)S;

  cvt_bf16<<<(64 * 128 + 255) / 256, 256, 0, stream>>>(W0, wb0, 64 * 128);
  cvt_bf16<<<(64 * 64  + 255) / 256, 256, 0, stream>>>(W1, wb1, 64 * 64);
  cvt_bf16<<<(128 * 64 + 255) / 256, 256, 0, stream>>>(W2, wb2, 128 * 64);
  transpose_pts<<<Bv * (Nv / 32), 256, 0, stream>>>(points, xT);

  // layer 0: fused gather (center+neighbor, -1 pad) + GEMM 128->64
  gemm_bn<128, 64, true, false><<<NWG, 256, 0, stream>>>(
      nullptr, xT, nearIdx, wb0, b0, nullptr, nullptr, y01, prt);
  reduce_stats<<<1, 128, 0, stream>>>(prt, NWG, 64, invM, g0, e0, sc0, sh0);

  // layer 1: BN0+ReLU fused into A-load, GEMM 64->64 (in place: row s -> row s)
  gemm_bn<64, 64, false, true><<<NWG, 256, 0, stream>>>(
      y01, nullptr, nullptr, wb1, b1, sc0, sh0, y01, prt);
  reduce_stats<<<1, 128, 0, stream>>>(prt, NWG, 64, invM, g1, e1, sc1, sh1);

  // layer 2: BN1+ReLU fused, GEMM 64->128
  gemm_bn<64, 128, false, true><<<NWG, 256, 0, stream>>>(
      y01, nullptr, nullptr, wb2, b2, sc1, sh1, y2, prt);
  reduce_stats<<<1, 128, 0, stream>>>(prt, NWG, 128, invM, g2, e2, sc2, sh2);

  // BN2+ReLU + neighbor max + channel max
  out_kernel<<<Bv * (Nv / 32), 256, 0, stream>>>(y2, sc2, sh2, out);
}
